// ProbSparseSelfAttention_42777874268840
// MI455X (gfx1250) — compile-verified, matrix operand order checked by reading, not run on hardware
//
#include <hip/hip_runtime.h>
#include <hip/hip_bf16.h>

// ---------------------------------------------------------------------------
// Self attention for MI455X (gfx1250, wave32, WMMA).
// B=4, S=4096, D_MODEL=128, N_HEAD=2, HEAD_DIM=64, SCALE=1/8, NEG_INF=-1e9.
//
//   - f32 projections (QKV, output) for accuracy; attention core in f16 WMMA
//     with f32 accumulation (v_wmma_f32_16x16x32_f16).
//   - Flash attention computing S^T = K * Q^T: softmax'ed scores land
//     directly in the B-fragment layout needed for O^T = V^T * P^T
//     (zero shuffles / LDS for the P matrix).
//   - Softmax in log2 domain: SCALE*log2(e) folded into Q, probabilities via
//     raw v_exp_f32 (__builtin_amdgcn_exp2f) — no per-element mul by log2e.
//   - Per-query stats live in lanes (n, n+16): local reduce + one shfl_xor.
// ---------------------------------------------------------------------------

typedef __attribute__((ext_vector_type(16))) _Float16 v16h;
typedef __attribute__((ext_vector_type(8)))  _Float16 v8h;
typedef __attribute__((ext_vector_type(8)))  float    v8f;

#define B_SZ   4
#define S_LEN  4096
#define DMODEL 128
#define NHEAD  2
#define HDIM   64
#define SCALE  0.125f
#define NEGINF -1e9f
#define LOG2E  1.4426950408889634f

static __device__ inline v16h frag_combine(v8h lo, v8h hi) {
  return __builtin_shufflevector(lo, hi, 0,1,2,3,4,5,6,7,8,9,10,11,12,13,14,15);
}

// Load a 16-bit A/B-style fragment: 8 contiguous f16 at p, 8 more at p+16.
static __device__ inline v16h frag_load(const _Float16* __restrict__ p) {
  v8h lo = *(const v8h*)(p);
  v8h hi = *(const v8h*)(p + 16);
  return frag_combine(lo, hi);
}

// ---------------------------------------------------------------------------
// Kernel 0: transpose a [rows, cols] f32 matrix into [cols, rows].
// ---------------------------------------------------------------------------
__global__ __launch_bounds__(256) void transpose_kernel(
    const float* __restrict__ src, float* __restrict__ dst, int rows, int cols) {
  int idx = blockIdx.x * 256 + threadIdx.x;
  if (idx >= rows * cols) return;
  int r = idx / cols, c = idx % cols;
  dst[c * rows + r] = src[idx];
}

// ---------------------------------------------------------------------------
// Kernel 1: key_mask (bool/u8) -> additive f32 mask (0 keep, -1e9 drop).
// (-1e9 in the log2-domain scores still underflows exp2 to exactly 0.)
// ---------------------------------------------------------------------------
__global__ __launch_bounds__(256) void mask_kernel(
    const unsigned char* __restrict__ mask, float* __restrict__ madd) {
  int idx = blockIdx.x * 256 + threadIdx.x;
  if (idx >= B_SZ * NHEAD * S_LEN) return;
  madd[idx] = mask[idx] ? 0.0f : NEGINF;
}

// ---------------------------------------------------------------------------
// Kernel 2: QKV projection (f32 math), scatter into f16 head layouts.
//   Qh [BH, S, 64]   (pre-scaled by SCALE * LOG2E -> scores in log2 domain)
//   Kh [BH, S, 64]
//   VhT[BH, 64, S]   (transposed: dim-major, contiguous along keys)
// ---------------------------------------------------------------------------
__global__ __launch_bounds__(256) void qkv_kernel(
    const float* __restrict__ x, const float* __restrict__ wT,
    const float* __restrict__ bias,
    _Float16* __restrict__ Qh, _Float16* __restrict__ Kh,
    _Float16* __restrict__ VhT) {
  int idx = blockIdx.x * 256 + threadIdx.x;
  if (idx >= B_SZ * S_LEN * 3 * DMODEL) return;
  int row = idx / (3 * DMODEL);          // b*S + s
  int col = idx % (3 * DMODEL);
  const float4* xr = (const float4*)(x + (size_t)row * DMODEL);
  const float4* wr = (const float4*)(wT + (size_t)col * DMODEL);
  float acc = 0.0f;
#pragma unroll
  for (int i = 0; i < DMODEL / 4; ++i) {
    float4 a = xr[i], w = wr[i];
    acc = fmaf(a.x, w.x, acc); acc = fmaf(a.y, w.y, acc);
    acc = fmaf(a.z, w.z, acc); acc = fmaf(a.w, w.w, acc);
  }
  acc += bias[col];

  int seg = col / DMODEL;                // 0=Q, 1=K, 2=V
  int c   = col % DMODEL;
  int h   = c / HDIM;
  int d   = c % HDIM;
  int b   = row / S_LEN;
  int s   = row % S_LEN;
  size_t bh = (size_t)b * NHEAD + h;
  if (seg == 0)      Qh[(bh * S_LEN + s) * HDIM + d] = (_Float16)(acc * (SCALE * LOG2E));
  else if (seg == 1) Kh[(bh * S_LEN + s) * HDIM + d] = (_Float16)acc;
  else               VhT[(bh * HDIM + d) * S_LEN + s] = (_Float16)acc;
}

// ---------------------------------------------------------------------------
// Kernel 3: flash attention core.
// grid = (S/128, B*H), block = 128 (4 waves). Each wave: 32 queries (2 tiles
// of 16), streams key blocks of 32. Per block-iter: 8 WMMA for S^T, 8 WMMA
// for O^T accumulation.
// ---------------------------------------------------------------------------
#define QT 2   // query tiles (of 16) per wave

__global__ __launch_bounds__(128) void attn_kernel(
    const _Float16* __restrict__ Qh, const _Float16* __restrict__ Kh,
    const _Float16* __restrict__ VhT, const float* __restrict__ madd,
    float* __restrict__ ctx) {
  const int bh   = blockIdx.y;                 // 0 .. B*H-1
  const int b    = bh / NHEAD;
  const int h    = bh % NHEAD;
  const int lane = threadIdx.x & 31;
  const int wave = threadIdx.x >> 5;
  const int n    = lane & 15;                  // query column / row-in-tile
  const int hi   = lane >> 4;                  // half-wave selector
  const int qbase = blockIdx.x * (4 * QT * 16) + wave * (QT * 16);

  const _Float16* qb = Qh  + (size_t)bh * S_LEN * HDIM;
  const _Float16* kp = Kh  + (size_t)bh * S_LEN * HDIM;
  const _Float16* vp = VhT + (size_t)bh * HDIM * S_LEN;
  const float*    mp = madd + (size_t)bh * S_LEN;

  // Q^T B-fragments: lane = query n, elements = head dims (pattern order).
  v16h qf[QT][2];
#pragma unroll
  for (int qt = 0; qt < QT; ++qt)
#pragma unroll
    for (int ks = 0; ks < 2; ++ks)
      qf[qt][ks] = frag_load(qb + (size_t)(qbase + qt * 16 + n) * HDIM
                                + ks * 32 + hi * 8);

  v8f   acc[QT][4];
  float m_i[QT], l_i[QT];
#pragma unroll
  for (int qt = 0; qt < QT; ++qt) {
    m_i[qt] = -1e30f; l_i[qt] = 0.0f;
#pragma unroll
    for (int dt = 0; dt < 4; ++dt)
#pragma unroll
      for (int i = 0; i < 8; ++i) acc[qt][dt][i] = 0.0f;
  }

  for (int kb = 0; kb < S_LEN; kb += 32) {
    // K A-fragments: lane = key row, elements = head dims.
    v16h kf[2][2];
#pragma unroll
    for (int kt = 0; kt < 2; ++kt)
#pragma unroll
      for (int ks = 0; ks < 2; ++ks)
        kf[kt][ks] = frag_load(kp + (size_t)(kb + kt * 16 + n) * HDIM
                                  + ks * 32 + hi * 8);
    // V^T A-fragments: lane = dim row, elements = keys.
    v16h vf[4];
#pragma unroll
    for (int dt = 0; dt < 4; ++dt)
      vf[dt] = frag_load(vp + (size_t)(dt * 16 + n) * S_LEN + kb + hi * 8);

    // Additive key mask, matching S^T C-layout element order.
    const float* mq = mp + kb + hi * 8;
    float4 a0 = *(const float4*)(mq);      float4 a1 = *(const float4*)(mq + 4);
    float4 b0 = *(const float4*)(mq + 16); float4 b1 = *(const float4*)(mq + 20);
    float mk0[8] = {a0.x, a0.y, a0.z, a0.w, a1.x, a1.y, a1.z, a1.w};
    float mk1[8] = {b0.x, b0.y, b0.z, b0.w, b1.x, b1.y, b1.z, b1.w};

#pragma unroll
    for (int qt = 0; qt < QT; ++qt) {
      // S^T = K * Q^T  (two 16x16 key tiles, K-dim = head dim, 2 steps of 32)
      v8f s0, s1;
#pragma unroll
      for (int i = 0; i < 8; ++i) { s0[i] = 0.0f; s1[i] = 0.0f; }
      s0 = __builtin_amdgcn_wmma_f32_16x16x32_f16(false, kf[0][0], false,
             qf[qt][0], (short)0, s0, false, false);
      s0 = __builtin_amdgcn_wmma_f32_16x16x32_f16(false, kf[0][1], false,
             qf[qt][1], (short)0, s0, false, false);
      s1 = __builtin_amdgcn_wmma_f32_16x16x32_f16(false, kf[1][0], false,
             qf[qt][0], (short)0, s1, false, false);
      s1 = __builtin_amdgcn_wmma_f32_16x16x32_f16(false, kf[1][1], false,
             qf[qt][1], (short)0, s1, false, false);

#pragma unroll
      for (int i = 0; i < 8; ++i) { s0[i] += mk0[i]; s1[i] += mk1[i]; }

      // Online softmax (log2 domain): this lane holds 16 of query n's 32
      // scores; the complementary 16 live in lane n^16.
      float mx = s0[0];
#pragma unroll
      for (int i = 1; i < 8; ++i) mx = fmaxf(mx, s0[i]);
#pragma unroll
      for (int i = 0; i < 8; ++i) mx = fmaxf(mx, s1[i]);
      mx = fmaxf(mx, __shfl_xor(mx, 16, 32));
      float mn   = fmaxf(m_i[qt], mx);
      float corr = __builtin_amdgcn_exp2f(m_i[qt] - mn);

      float pe[16]; float se = 0.0f;
#pragma unroll
      for (int i = 0; i < 8; ++i) {
        pe[i] = __builtin_amdgcn_exp2f(s0[i] - mn); se += pe[i];
      }
#pragma unroll
      for (int i = 0; i < 8; ++i) {
        pe[8 + i] = __builtin_amdgcn_exp2f(s1[i] - mn); se += pe[8 + i];
      }
      se += __shfl_xor(se, 16, 32);
      l_i[qt] = l_i[qt] * corr + se;
      m_i[qt] = mn;

      // P^T B-fragment: exp2(S^T) is already in B-fragment element order.
      v16h pf;
#pragma unroll
      for (int i = 0; i < 16; ++i) pf[i] = (_Float16)pe[i];

      // O^T += V^T * P^T  (K-dim = 32 keys), after rescaling by corr.
#pragma unroll
      for (int dt = 0; dt < 4; ++dt) {
#pragma unroll
        for (int i = 0; i < 8; ++i) acc[qt][dt][i] *= corr;
        acc[qt][dt] = __builtin_amdgcn_wmma_f32_16x16x32_f16(false, vf[dt],
                        false, pf, (short)0, acc[qt][dt], false, false);
      }
    }
  }

  // Epilogue: O[query][dim] = acc / l.  Lane owns query n, dims dt*16+8*hi+v.
#pragma unroll
  for (int qt = 0; qt < QT; ++qt) {
    float inv = 1.0f / fmaxf(l_i[qt], 1e-37f);
    int s = qbase + qt * 16 + n;
    float* op = ctx + ((size_t)b * S_LEN + s) * DMODEL + h * HDIM;
#pragma unroll
    for (int dt = 0; dt < 4; ++dt) {
      float* o = op + dt * 16 + hi * 8;
      float4 o0 = {acc[qt][dt][0] * inv, acc[qt][dt][1] * inv,
                   acc[qt][dt][2] * inv, acc[qt][dt][3] * inv};
      float4 o1 = {acc[qt][dt][4] * inv, acc[qt][dt][5] * inv,
                   acc[qt][dt][6] * inv, acc[qt][dt][7] * inv};
      *(float4*)(o)     = o0;
      *(float4*)(o + 4) = o1;
    }
  }
}

// ---------------------------------------------------------------------------
// Kernel 4: output projection (f32): out = ctx @ w_proj + b_proj.
// ---------------------------------------------------------------------------
__global__ __launch_bounds__(256) void proj_kernel(
    const float* __restrict__ ctx, const float* __restrict__ wT,
    const float* __restrict__ bias, float* __restrict__ out) {
  int idx = blockIdx.x * 256 + threadIdx.x;
  if (idx >= B_SZ * S_LEN * DMODEL) return;
  int row = idx / DMODEL, col = idx % DMODEL;
  const float4* xr = (const float4*)(ctx + (size_t)row * DMODEL);
  const float4* wr = (const float4*)(wT + (size_t)col * DMODEL);
  float acc = 0.0f;
#pragma unroll
  for (int i = 0; i < DMODEL / 4; ++i) {
    float4 a = xr[i], w = wr[i];
    acc = fmaf(a.x, w.x, acc); acc = fmaf(a.y, w.y, acc);
    acc = fmaf(a.z, w.z, acc); acc = fmaf(a.w, w.w, acc);
  }
  out[idx] = acc + bias[col];
}

// ---------------------------------------------------------------------------
extern "C" void kernel_launch(void* const* d_in, const int* in_sizes, int n_in,
                              void* d_out, int out_size, void* d_ws, size_t ws_size,
                              hipStream_t stream) {
  const float*         x      = (const float*)d_in[0];
  const unsigned char* mask   = (const unsigned char*)d_in[1];
  const float*         w_qkv  = (const float*)d_in[2];
  const float*         b_qkv  = (const float*)d_in[3];
  const float*         w_proj = (const float*)d_in[4];
  const float*         b_proj = (const float*)d_in[5];
  float*               out    = (float*)d_out;

  // Workspace carve-up (all offsets multiples of 256B).
  char* w = (char*)d_ws;
  float* wqkvT  = (float*)w;  w += (size_t)384 * 128 * 4;          // 192 KB
  float* wprojT = (float*)w;  w += (size_t)128 * 128 * 4;          //  64 KB
  _Float16* Qh  = (_Float16*)w; w += (size_t)B_SZ * NHEAD * S_LEN * HDIM * 2; // 4 MB
  _Float16* Kh  = (_Float16*)w; w += (size_t)B_SZ * NHEAD * S_LEN * HDIM * 2; // 4 MB
  _Float16* VhT = (_Float16*)w; w += (size_t)B_SZ * NHEAD * HDIM * S_LEN * 2; // 4 MB
  float* madd   = (float*)w;  w += (size_t)B_SZ * NHEAD * S_LEN * 4;          // 128 KB
  float* ctx    = (float*)w;  w += (size_t)B_SZ * S_LEN * DMODEL * 4;         // 8 MB

  transpose_kernel<<<(128 * 384 + 255) / 256, 256, 0, stream>>>(w_qkv, wqkvT, 128, 384);
  transpose_kernel<<<(128 * 128 + 255) / 256, 256, 0, stream>>>(w_proj, wprojT, 128, 128);
  mask_kernel<<<(B_SZ * NHEAD * S_LEN + 255) / 256, 256, 0, stream>>>(mask, madd);
  qkv_kernel<<<(B_SZ * S_LEN * 3 * DMODEL + 255) / 256, 256, 0, stream>>>(
      x, wqkvT, b_qkv, Qh, Kh, VhT);
  attn_kernel<<<dim3(S_LEN / 128, B_SZ * NHEAD), 128, 0, stream>>>(
      Qh, Kh, VhT, madd, ctx);
  proj_kernel<<<(B_SZ * S_LEN * DMODEL + 255) / 256, 256, 0, stream>>>(
      ctx, wprojT, b_proj, out);
}
